// Attention_17300128268584
// MI455X (gfx1250) — matrix operand order repaired and checked by reading, not compiled, not run on hardware
//
#include <hip/hip_runtime.h>
#include <hip/hip_bf16.h>

typedef __attribute__((ext_vector_type(16))) __bf16 v16bf;
typedef __attribute__((ext_vector_type(8)))  float  v8f;
typedef __attribute__((ext_vector_type(16))) unsigned short v16u;

struct U4 { unsigned int x, y, z, w; };

union FragU {
    v16u v;
    unsigned short s[16];
    U4 q[2];
};

__device__ __forceinline__ unsigned short f2bf(float f) {
    unsigned int u = __float_as_uint(f);
    u += 0x7FFFu + ((u >> 16) & 1u);   // round-to-nearest-even
    return (unsigned short)(u >> 16);
}
__device__ __forceinline__ float bf2f(unsigned short s) {
    return __uint_as_float(((unsigned int)s) << 16);
}
__device__ __forceinline__ v8f v8f_zero() {
    v8f z = {0.f, 0.f, 0.f, 0.f, 0.f, 0.f, 0.f, 0.f};
    return z;
}
__device__ __forceinline__ v8f wmma_bf16(const FragU& a, const FragU& b, v8f c) {
    return __builtin_amdgcn_wmma_f32_16x16x32_bf16(
        false, __builtin_bit_cast(v16bf, a.v),
        false, __builtin_bit_cast(v16bf, b.v),
        (short)0, c, false, false);
}

// ---------------- sizes ----------------
// B=16, C=128, IH=IW=32, N=1024, H=4, D=32
#define NB   16
#define NC   128
#define NPIX 1024
#define NH   4
#define ND   32

// ---------------- prep kernels ----------------
__global__ void k_zero_stats(float* stats) {
    int i = threadIdx.x;
    if (i < 96) stats[i] = 0.f;
}

// x [b][c][h][w] f32 -> x_t [b][h][w][c] bf16 (channel-contiguous for LDS staging)
__global__ void k_prep_x(const float* __restrict__ x, unsigned short* __restrict__ xt, int n) {
    int i = blockIdx.x * 256 + threadIdx.x;
    if (i >= n) return;
    int c = i & 127;
    int pix = (i >> 7) & 1023;
    int b = i >> 17;
    xt[i] = f2bf(x[((size_t)(b * NC + c)) * NPIX + pix]);
}

// w[pr][oc][ic][ky][kx] (f32) -> wb[pr][tap][oc][ic] (bf16)
__global__ void k_prep_w(const float* __restrict__ wq, const float* __restrict__ wk,
                         const float* __restrict__ wv, unsigned short* __restrict__ wb) {
    int i = blockIdx.x * 256 + threadIdx.x;
    const int per = NC * NC * 9;
    if (i >= 3 * per) return;
    int pr = i / per, r = i % per;
    int oc = r / (NC * 9);
    int r2 = r % (NC * 9);
    int ic = r2 / 9;
    int tap = r2 % 9;
    const float* w = (pr == 0) ? wq : (pr == 1) ? wk : wv;
    wb[(((size_t)(pr * 9 + tap)) * NC + oc) * NC + ic] = f2bf(w[(oc * NC + ic) * 9 + tap]);
}

__global__ void k_prep_w1(const float* __restrict__ w1, unsigned short* __restrict__ wb) {
    int i = blockIdx.x * 256 + threadIdx.x;
    if (i < NC * NC) wb[i] = f2bf(w1[i]);
}

// ---------------- conv3x3 as implicit GEMM (WMMA) ----------------
// grid (64 pixel-tiles, 16 batch, 3 proj), block 128 (4 waves x 2 oc-tiles)
__global__ __launch_bounds__(128)
void k_conv(const unsigned short* __restrict__ xtg,  // [b][h][w][c] bf16
            const unsigned short* __restrict__ wb,   // [pr][tap][oc][ic] bf16
            unsigned short* __restrict__ cout,       // [pr][b][oc][p] bf16
            float* __restrict__ stats)               // [pr][b][2] f32 (sum, sumsq)
{
    __shared__ unsigned short xt[54 * 128];          // [3 rows][18 cols][128 ic]
    const int p0 = blockIdx.x * 16;
    const int b  = blockIdx.y;
    const int pr = blockIdx.z;
    const int py = p0 >> 5, px0 = p0 & 31;
    const int t = threadIdx.x;

    // cooperative halo staging: 16 threads per spatial position, 8-channel chunks,
    // b128 global load -> b128 LDS store
    {
        const int icb = (t & 15) * 8;
        for (int pos = t >> 4; pos < 54; pos += 8) {
            int rr = pos / 18, cc2 = pos % 18;
            int row = py + rr - 1;
            int col = px0 + cc2 - 1;
            U4 v = {0u, 0u, 0u, 0u};
            if (row >= 0 && row < 32 && col >= 0 && col < 32) {
                v = *(const U4*)(xtg + (((size_t)(b * 32 + row)) * 32 + col) * NC + icb);
            }
            *(U4*)(xt + pos * 128 + icb) = v;
        }
    }
    __syncthreads();

    const int lane = t & 31, wid = t >> 5;
    const int hi = lane >> 4, ln = lane & 15;
    v8f acc0 = v8f_zero(), acc1 = v8f_zero();
    const int oct0 = wid * 2, oct1 = wid * 2 + 1;

    for (int tap = 0; tap < 9; ++tap) {
        int dy = tap / 3, dx = tap % 3;
        const unsigned short* wtap = wb + ((size_t)(pr * 9 + tap)) * NC * NC;
        const unsigned short* xrow = xt + (dy * 18 + ln + dx) * 128 + 16 * hi;
        #pragma unroll
        for (int cc = 0; cc < 4; ++cc) {
            FragU bx;                                   // B: (ic chunk 32) x (16 pixels)
            bx.q[0] = *(const U4*)(xrow + cc * 32);     // ds_load_b128
            bx.q[1] = *(const U4*)(xrow + cc * 32 + 8); // ds_load_b128
            const unsigned short* wa0 = wtap + (size_t)(oct0 * 16 + ln) * NC + cc * 32;
            const unsigned short* wa1 = wtap + (size_t)(oct1 * 16 + ln) * NC + cc * 32;
            FragU a0, a1;
            a0.q[0] = *(const U4*)(wa0 + 16 * hi);
            a0.q[1] = *(const U4*)(wa0 + 16 * hi + 8);
            a1.q[0] = *(const U4*)(wa1 + 16 * hi);
            a1.q[1] = *(const U4*)(wa1 + 16 * hi + 8);
            acc0 = wmma_bf16(a0, bx, acc0);
            acc1 = wmma_bf16(a1, bx, acc1);
        }
    }

    // GroupNorm stats (pre-rounding, f32 exact values)
    float s = 0.f, s2 = 0.f;
    #pragma unroll
    for (int r = 0; r < 8; ++r) {
        s  += acc0[r] + acc1[r];
        s2 += acc0[r] * acc0[r] + acc1[r] * acc1[r];
    }
    #pragma unroll
    for (int m = 1; m < 32; m <<= 1) {
        s  += __shfl_xor(s,  m, 32);
        s2 += __shfl_xor(s2, m, 32);
    }
    if (lane == 0) {
        atomicAdd(&stats[(pr * NB + b) * 2 + 0], s);
        atomicAdd(&stats[(pr * NB + b) * 2 + 1], s2);
    }

    unsigned short* dst = cout + ((size_t)(pr * NB + b)) * NC * NPIX;
    #pragma unroll
    for (int r = 0; r < 8; ++r) {
        int oc0 = oct0 * 16 + r + 8 * hi;
        int oc1 = oct1 * 16 + r + 8 * hi;
        dst[(size_t)oc0 * NPIX + p0 + ln] = f2bf(acc0[r]);
        dst[(size_t)oc1 * NPIX + p0 + ln] = f2bf(acc1[r]);
    }
}

__global__ void k_stats(const float* __restrict__ stats, float* __restrict__ mr) {
    int i = threadIdx.x;
    if (i < 48) {
        const float cnt = (float)(NC * NPIX);
        float s = stats[i * 2], s2 = stats[i * 2 + 1];
        float mean = s / cnt;
        float var = s2 / cnt - mean * mean;
        mr[i * 2]     = mean;
        mr[i * 2 + 1] = rsqrtf(var + 1e-6f);
    }
}

// ---------------- GroupNorm + GELU + head reshape ----------------
__global__ __launch_bounds__(256)
void k_gn(const unsigned short* __restrict__ cb, const float* __restrict__ mr,
          const float* __restrict__ gq, const float* __restrict__ bq,
          const float* __restrict__ gk, const float* __restrict__ bk,
          const float* __restrict__ gv, const float* __restrict__ bv,
          unsigned short* __restrict__ qh,   // [b][h][n][d]
          unsigned short* __restrict__ kh,   // [b][h][n][d]
          unsigned short* __restrict__ vt)   // [b][h][d][n]
{
    int i = blockIdx.x * 256 + threadIdx.x;
    if (i >= 3 * NB * NC * NPIX) return;
    int p = i & 1023, c = (i >> 10) & 127, b = (i >> 17) & 15, pr = i >> 21;
    float v = bf2f(cb[i]);
    float mean = mr[(pr * NB + b) * 2], rstd = mr[(pr * NB + b) * 2 + 1];
    const float* g  = (pr == 0) ? gq : (pr == 1) ? gk : gv;
    const float* bt = (pr == 0) ? bq : (pr == 1) ? bk : bv;
    float xn = (v - mean) * rstd * g[c] + bt[c];
    float ge = 0.5f * xn * (1.f + erff(xn * 0.70710678118f));  // exact GELU
    unsigned short o = f2bf(ge);
    int h = c >> 5, d = c & 31;
    if (pr == 0)      qh[((size_t)(b * NH + h) * NPIX + p) * ND + d] = o;
    else if (pr == 1) kh[((size_t)(b * NH + h) * NPIX + p) * ND + d] = o;
    else              vt[((size_t)(b * NH + h) * ND + d) * NPIX + p] = o;
}

// ---------------- flash attention (WMMA, online softmax) ----------------
// grid (8 q-blocks, 4 heads, 16 batch), block 256 = 8 waves; 1 wave = 16 queries
__global__ __launch_bounds__(256)
void k_attn(const unsigned short* __restrict__ qh,
            const unsigned short* __restrict__ kh,
            const unsigned short* __restrict__ vt,
            const float* __restrict__ table,   // [(2*32-1)^2][4]
            const int*   __restrict__ rel,     // [N*N]
            unsigned short* __restrict__ ab)   // [b][n][c] bf16
{
    __shared__ unsigned short pbuf[8 * 16 * 32];
    const int t = threadIdx.x, lane = t & 31, wid = t >> 5;
    const int hi = lane >> 4, ln = lane & 15;
    const int b = blockIdx.z, h = blockIdx.y;
    const int q0 = (blockIdx.x * 8 + wid) * 16;
    const size_t bh = (size_t)(b * NH + h);

    FragU aq;   // Q tile 16x32 (A layout)
    {
        const unsigned short* qp = qh + (bh * NPIX + q0 + ln) * ND;
        aq.q[0] = *(const U4*)(qp + 16 * hi);
        aq.q[1] = *(const U4*)(qp + 16 * hi + 8);
    }
    float mrow[8], lrow[8];
    v8f o0 = v8f_zero(), o1 = v8f_zero();
    #pragma unroll
    for (int r = 0; r < 8; ++r) { mrow[r] = -1e30f; lrow[r] = 0.f; }
    unsigned short* pw = pbuf + wid * 512;

    for (int m0 = 0; m0 < NPIX; m0 += 32) {
        // K^T fragments: B(kdim, key) = k[key][kdim], contiguous in d per lane
        FragU bk0, bk1;
        {
            const unsigned short* kp = kh + (bh * NPIX + m0 + ln) * ND + 16 * hi;
            bk0.q[0] = *(const U4*)kp; bk0.q[1] = *(const U4*)(kp + 8);
        }
        {
            const unsigned short* kp = kh + (bh * NPIX + m0 + 16 + ln) * ND + 16 * hi;
            bk1.q[0] = *(const U4*)kp; bk1.q[1] = *(const U4*)(kp + 8);
        }
        v8f s0 = wmma_bf16(aq, bk0, v8f_zero());
        v8f s1 = wmma_bf16(aq, bk1, v8f_zero());

        // relative-position bias
        #pragma unroll
        for (int r = 0; r < 8; ++r) {
            int n = q0 + r + 8 * hi;
            const int* rp = rel + (size_t)n * NPIX + m0;
            s0[r] += table[rp[ln] * NH + h];
            s1[r] += table[rp[16 + ln] * NH + h];
        }

        // online softmax (row = across 16 lanes of a half-wave)
        #pragma unroll
        for (int r = 0; r < 8; ++r) {
            float mx = fmaxf(s0[r], s1[r]);
            mx = fmaxf(mx, __shfl_xor(mx, 1, 32));
            mx = fmaxf(mx, __shfl_xor(mx, 2, 32));
            mx = fmaxf(mx, __shfl_xor(mx, 4, 32));
            mx = fmaxf(mx, __shfl_xor(mx, 8, 32));
            float mnew = fmaxf(mrow[r], mx);
            float sc   = __expf(mrow[r] - mnew);
            float p0v  = __expf(s0[r] - mnew);
            float p1v  = __expf(s1[r] - mnew);
            float rs = p0v + p1v;
            rs += __shfl_xor(rs, 1, 32);
            rs += __shfl_xor(rs, 2, 32);
            rs += __shfl_xor(rs, 4, 32);
            rs += __shfl_xor(rs, 8, 32);
            lrow[r] = lrow[r] * sc + rs;
            mrow[r] = mnew;
            o0[r] *= sc; o1[r] *= sc;
            pw[(r + 8 * hi) * 32 + ln]      = f2bf(p0v);   // C-layout -> LDS
            pw[(r + 8 * hi) * 32 + 16 + ln] = f2bf(p1v);
        }
        asm volatile("" ::: "memory");
        __builtin_amdgcn_wave_barrier();               // same-wave DS ops are in-order

        FragU ap;   // P tile 16x32 re-read in A layout
        {
            const unsigned short* pp = pw + ln * 32;
            ap.q[0] = *(const U4*)(pp + 16 * hi);
            ap.q[1] = *(const U4*)(pp + 16 * hi + 8);
        }
        FragU bv0f, bv1f;  // V fragments from transposed V: contiguous keys per lane
        {
            const unsigned short* vp = vt + (bh * ND + ln) * NPIX + m0 + 16 * hi;
            bv0f.q[0] = *(const U4*)vp; bv0f.q[1] = *(const U4*)(vp + 8);
        }
        {
            const unsigned short* vp = vt + (bh * ND + 16 + ln) * NPIX + m0 + 16 * hi;
            bv1f.q[0] = *(const U4*)vp; bv1f.q[1] = *(const U4*)(vp + 8);
        }
        o0 = wmma_bf16(ap, bv0f, o0);
        o1 = wmma_bf16(ap, bv1f, o1);

        if (m0 + 32 < NPIX) {
            __builtin_prefetch(kh + (bh * NPIX + m0 + 32 + ln) * ND, 0, 1);
            __builtin_prefetch(vt + (bh * ND + ln) * NPIX + m0 + 32, 0, 1);
        }
    }

    #pragma unroll
    for (int r = 0; r < 8; ++r) {
        float inv = 1.f / lrow[r];
        int n = q0 + r + 8 * hi;
        unsigned short* op = ab + ((size_t)(b * NPIX + n)) * NC + h * ND;
        op[ln]      = f2bf(o0[r] * inv);
        op[16 + ln] = f2bf(o1[r] * inv);
    }
}

// ---------------- 1x1 conv epilogue (WMMA GEMM + bias) ----------------
// grid (64 pixel-tiles, 16 batch), block 128 (4 waves x 2 oc-tiles)
__global__ __launch_bounds__(128)
void k_out(const unsigned short* __restrict__ ab,   // [b][p][c] bf16
           const unsigned short* __restrict__ w1,   // [oc][c] bf16
           const float* __restrict__ outb,
           float* __restrict__ out)                 // [b][oc][p] f32
{
    const int t = threadIdx.x, lane = t & 31, wid = t >> 5;
    const int hi = lane >> 4, ln = lane & 15;
    const int p0 = blockIdx.x * 16, b = blockIdx.y;
    v8f c0 = v8f_zero(), c1 = v8f_zero();
    const int oct0 = wid * 2, oct1 = wid * 2 + 1;
    #pragma unroll
    for (int cc = 0; cc < 4; ++cc) {
        FragU bf;
        const unsigned short* bp = ab + ((size_t)(b * NPIX + p0 + ln)) * NC + cc * 32 + 16 * hi;
        bf.q[0] = *(const U4*)bp; bf.q[1] = *(const U4*)(bp + 8);
        FragU a0, a1;
        const unsigned short* wp0 = w1 + (size_t)(oct0 * 16 + ln) * NC + cc * 32;
        const unsigned short* wp1 = w1 + (size_t)(oct1 * 16 + ln) * NC + cc * 32;
        a0.q[0] = *(const U4*)(wp0 + 16 * hi); a0.q[1] = *(const U4*)(wp0 + 16 * hi + 8);
        a1.q[0] = *(const U4*)(wp1 + 16 * hi); a1.q[1] = *(const U4*)(wp1 + 16 * hi + 8);
        c0 = wmma_bf16(a0, bf, c0);
        c1 = wmma_bf16(a1, bf, c1);
    }
    #pragma unroll
    for (int r = 0; r < 8; ++r) {
        int oc0 = oct0 * 16 + r + 8 * hi;
        int oc1 = oct1 * 16 + r + 8 * hi;
        out[((size_t)(b * NC + oc0)) * NPIX + p0 + ln] = c0[r] + outb[oc0];
        out[((size_t)(b * NC + oc1)) * NPIX + p0 + ln] = c1[r] + outb[oc1];
    }
}

// ---------------- launcher ----------------
extern "C" void kernel_launch(void* const* d_in, const int* in_sizes, int n_in,
                              void* d_out, int out_size, void* d_ws, size_t ws_size,
                              hipStream_t stream) {
    const float* x     = (const float*)d_in[0];
    const float* wq    = (const float*)d_in[1];
    const float* gq    = (const float*)d_in[2];
    const float* bq    = (const float*)d_in[3];
    const float* wk    = (const float*)d_in[4];
    const float* gk    = (const float*)d_in[5];
    const float* bk    = (const float*)d_in[6];
    const float* wv    = (const float*)d_in[7];
    const float* gv    = (const float*)d_in[8];
    const float* bv    = (const float*)d_in[9];
    const float* table = (const float*)d_in[10];
    const int*   rel   = (const int*)d_in[11];
    const float* out_w = (const float*)d_in[12];
    const float* out_b = (const float*)d_in[13];
    float* out = (float*)d_out;

    // workspace carve-up (all bf16 unless noted)
    char* ws = (char*)d_ws;
    size_t off = 0;
    auto take = [&](size_t bytes) { char* p = ws + off; off = (off + bytes + 255) & ~(size_t)255; return p; };
    unsigned short* xtg   = (unsigned short*)take((size_t)NB * NC * NPIX * 2);        // 4 MB
    unsigned short* wb    = (unsigned short*)take((size_t)3 * 9 * NC * NC * 2);       // 0.84 MB
    unsigned short* w1b   = (unsigned short*)take((size_t)NC * NC * 2);               // 32 KB
    float*          stats = (float*)take(96 * 4);
    float*          mr    = (float*)take(96 * 4);
    unsigned short* cb    = (unsigned short*)take((size_t)3 * NB * NC * NPIX * 2);    // 12 MB
    unsigned short* qh    = (unsigned short*)take((size_t)NB * NC * NPIX * 2);        // 4 MB
    unsigned short* kh    = (unsigned short*)take((size_t)NB * NC * NPIX * 2);        // 4 MB
    unsigned short* vt    = (unsigned short*)take((size_t)NB * NC * NPIX * 2);        // 4 MB
    unsigned short* abf   = (unsigned short*)take((size_t)NB * NC * NPIX * 2);        // 4 MB
    (void)ws_size; (void)n_in; (void)in_sizes; (void)out_size;

    k_zero_stats<<<1, 128, 0, stream>>>(stats);
    {
        int n = NB * NC * NPIX;
        k_prep_x<<<(n + 255) / 256, 256, 0, stream>>>(x, xtg, n);
    }
    k_prep_w<<<(3 * NC * NC * 9 + 255) / 256, 256, 0, stream>>>(wq, wk, wv, wb);
    k_prep_w1<<<(NC * NC + 255) / 256, 256, 0, stream>>>(out_w, w1b);

    k_conv<<<dim3(64, 16, 3), 128, 0, stream>>>(xtg, wb, cb, stats);
    k_stats<<<1, 64, 0, stream>>>(stats, mr);
    {
        int n = 3 * NB * NC * NPIX;
        k_gn<<<(n + 255) / 256, 256, 0, stream>>>(cb, mr, gq, bq, gk, bk, gv, bv, qh, kh, vt);
    }
    k_attn<<<dim3(8, 4, 16), 256, 0, stream>>>(qh, kh, vt, table, rel, abf);
    k_out<<<dim3(64, 16), 128, 0, stream>>>(abf, w1b, out_b, out);
}
